// ReferenceExperts_59502476918784
// MI455X (gfx1250) — matrix operand order repaired and checked by reading, not compiled, not run on hardware
//
#include <hip/hip_runtime.h>

#define TOK 4096   // B*S tokens
#define HID 1024   // H
#define DIM 2048   // D
#define NE  8      // experts

typedef __bf16 v16bf __attribute__((ext_vector_type(16)));
typedef __bf16 v8bf  __attribute__((ext_vector_type(8)));
typedef float  v8f   __attribute__((ext_vector_type(8)));

#define AP 40      // LDS row pitch in bf16 elements (80B: 16B aligned, odd bank stride)
#define BT 128     // token tile (M)
#define BN 128     // output-column tile (N)
#define KC 32      // K per WMMA step

static __device__ inline unsigned packbf2(float a, float b) {
  union { __bf16 h[2]; unsigned u; } t;
  t.h[0] = (__bf16)a; t.h[1] = (__bf16)b;
  return t.u;
}

// Load a 32-byte WMMA operand fragment as two b128 LDS reads.
static __device__ inline v16bf ld_frag(const __bf16* p, int second_off_elems) {
  union { v16bf v; v8bf h[2]; } t;
  t.h[0] = *(const v8bf*)p;
  t.h[1] = *(const v8bf*)(p + second_off_elems);
  return t.v;
}

static __device__ inline v8f wmma_bf16(v16bf a, v16bf b, v8f c) {
  return __builtin_amdgcn_wmma_f32_16x16x32_bf16(
      /*neg_a=*/false, a, /*neg_b=*/false, b,
      /*c_mod=*/(short)0, c, /*reuse_a=*/false, /*reuse_b=*/false);
}

// ---------------------------------------------------------------------------
// Kernel 1: gate_up GEMM + bias + clamp + GLU, writes bf16 act (TOK x DIM).
// Block: 256 threads = 8 waves (2 waves in M x 4 waves in N).
// Block tile: 128 tokens x 128 d-columns (each d has a gate col 2d, up col 2d+1).
// ---------------------------------------------------------------------------
__global__ __launch_bounds__(256) void moe_gateup_bf16(
    const float* __restrict__ x,          // TOK x HID
    const float* __restrict__ gup,        // NE x HID x 2*DIM
    const float* __restrict__ gub,        // NE x 2*DIM
    unsigned short* __restrict__ actw,    // TOK x DIM (bf16)
    int e)
{
  __shared__ __bf16 sA [2][BT * AP];   // x tile,   [m][k]
  __shared__ __bf16 sBg[2][BN * AP];   // gate W,   [n][k] (transposed)
  __shared__ __bf16 sBu[2][BN * AP];   // up W,     [n][k]
  __bf16* act = (__bf16*)actw;

  const int tid  = threadIdx.x;
  const int lane = tid & 31, wid = tid >> 5;
  const int wm = wid >> 2, wn = wid & 3;          // 2 x 4 wave grid
  const int t0 = blockIdx.x * BT;
  const int n0 = blockIdx.y * BN;                 // d-column base
  const float* W = gup + (size_t)e * HID * (2 * DIM);

  // global->LDS staging mapping
  const int ac = (tid & 7) * 4;                   // A: float4 column chunk
  const int ar = tid >> 3;                        // A: row 0..31 (x4 passes)
  const int bn = tid & 127;                       // B: n column
  const int bk = tid >> 7;                        // B: k-pair phase 0/1

  float4 aReg[4];
  float2 gReg[8], uReg[8];

  auto issueLoads = [&](int k0) {
    const float* xa = x + (size_t)t0 * HID + k0;
#pragma unroll
    for (int i = 0; i < 4; i++)
      aReg[i] = *(const float4*)(xa + (size_t)(ar + 32 * i) * HID + ac);
#pragma unroll
    for (int j = 0; j < 8; j++) {
      int p = bk + 2 * j;                                         // k-pair 0..15
      const float* wp = W + (size_t)(k0 + 2 * p) * (2 * DIM) + 2 * (n0 + bn);
      float2 r0 = *(const float2*)wp;                             // k=2p   : (gate, up)
      float2 r1 = *(const float2*)(wp + 2 * DIM);                 // k=2p+1 : (gate, up)
      gReg[j] = make_float2(r0.x, r1.x);
      uReg[j] = make_float2(r0.y, r1.y);
    }
  };
  auto storeLds = [&](int buf) {
#pragma unroll
    for (int i = 0; i < 4; i++) {
      unsigned* d = (unsigned*)&sA[buf][(ar + 32 * i) * AP + ac];
      d[0] = packbf2(aReg[i].x, aReg[i].y);
      d[1] = packbf2(aReg[i].z, aReg[i].w);
    }
#pragma unroll
    for (int j = 0; j < 8; j++) {
      int p = bk + 2 * j;
      *(unsigned*)&sBg[buf][bn * AP + 2 * p] = packbf2(gReg[j].x, gReg[j].y);
      *(unsigned*)&sBu[buf][bn * AP + 2 * p] = packbf2(uReg[j].x, uReg[j].y);
    }
  };

  const v8f vzero = {0.f, 0.f, 0.f, 0.f, 0.f, 0.f, 0.f, 0.f};
  v8f gacc[4][2], uacc[4][2];
#pragma unroll
  for (int m = 0; m < 4; m++)
#pragma unroll
    for (int n = 0; n < 2; n++) { gacc[m][n] = vzero; uacc[m][n] = vzero; }

  // WMMA fragment addressing (per documented gfx1250 layouts)
  const int fr = lane & 15;
  const int kA = (lane >> 4) * 8;    // A: lanes 16-31 hold K 8..15 / 24..31
  const int kB = (lane >> 4) * 8;    // B: lanes 16-31 hold K 8..15 / 24..31

  auto compute = [&](int buf) {
    v16bf bg[2], bu[2];
#pragma unroll
    for (int n = 0; n < 2; n++) {
      const __bf16* pg = &sBg[buf][(wn * 32 + n * 16 + fr) * AP + kB];
      bg[n] = ld_frag(pg, 16);
      const __bf16* pu = &sBu[buf][(wn * 32 + n * 16 + fr) * AP + kB];
      bu[n] = ld_frag(pu, 16);
    }
#pragma unroll
    for (int m = 0; m < 4; m++) {
      const __bf16* pa = &sA[buf][(wm * 64 + m * 16 + fr) * AP + kA];
      v16bf a = ld_frag(pa, 16);
#pragma unroll
      for (int n = 0; n < 2; n++) {
        gacc[m][n] = wmma_bf16(a, bg[n], gacc[m][n]);
        uacc[m][n] = wmma_bf16(a, bu[n], uacc[m][n]);
      }
    }
  };

  issueLoads(0); storeLds(0); __syncthreads();
  const int nk = HID / KC;  // 32
  for (int ks = 0; ks < nk; ks++) {
    int buf = ks & 1;
    if (ks + 1 < nk) issueLoads((ks + 1) * KC);
    compute(buf);
    if (ks + 1 < nk) storeLds(buf ^ 1);
    __syncthreads();
  }

  // Epilogue: bias + clamp + GLU, write bf16 intermediate.
  const float* gb = gub + (size_t)e * (2 * DIM);
#pragma unroll
  for (int m = 0; m < 4; m++) {
    int rbase = t0 + wm * 64 + m * 16 + ((lane >> 4) << 3);
#pragma unroll
    for (int n = 0; n < 2; n++) {
      int dcol = n0 + wn * 32 + n * 16 + fr;
      float bgv = gb[2 * dcol];
      float buv = gb[2 * dcol + 1];
#pragma unroll
      for (int v = 0; v < 8; v++) {
        float g = gacc[m][n][v] + bgv;
        float u = uacc[m][n][v] + buv;
        g = fminf(g, 7.0f);
        u = fminf(fmaxf(u, -7.0f), 7.0f);
        float glu = g / (1.0f + __expf(-1.702f * g));
        act[(size_t)(rbase + v) * DIM + dcol] = (__bf16)((u + 1.0f) * glu);
      }
    }
  }
}

// ---------------------------------------------------------------------------
// Kernel 2: down GEMM + bias + routing weight, accumulates into fp32 out.
// A tile is already bf16 -> fetched with async global->LDS copies (ASYNCcnt).
// ---------------------------------------------------------------------------
__global__ __launch_bounds__(256) void moe_down_bf16(
    const unsigned short* __restrict__ actw,  // TOK x DIM (bf16)
    const float* __restrict__ dw,             // NE x DIM x HID
    const float* __restrict__ db,             // NE x HID
    const float* __restrict__ rw,             // TOK x NE
    float* __restrict__ out,                  // TOK x HID
    int e)
{
  __shared__ __bf16 sA[2][BT * AP];   // act tile [m][k]
  __shared__ __bf16 sB[2][BN * AP];   // down W   [n][k] (transposed)
  const __bf16* act = (const __bf16*)actw;

  const int tid  = threadIdx.x;
  const int lane = tid & 31, wid = tid >> 5;
  const int wm = wid >> 2, wn = wid & 3;
  const int t0 = blockIdx.x * BT;
  const int n0 = blockIdx.y * BN;             // H column base
  const float* W = dw + (size_t)e * DIM * HID;

  const int ac = (tid & 3) * 8;               // A: 16B bf16 chunk
  const int ar = tid >> 2;                    // A: row 0..63 (x2 passes)
  const int bn = tid & 127;
  const int bk = tid >> 7;

  float2 bReg[8];

  auto issueA = [&](int k0, int buf) {
    const __bf16* ga = act + (size_t)(t0 + ar) * DIM + k0 + ac;
    unsigned l0 = (unsigned)(size_t)&sA[buf][ar * AP + ac];
    unsigned l1 = (unsigned)(size_t)&sA[buf][(ar + 64) * AP + ac];
    // async DMA of 16B per lane into LDS; tracked by ASYNCcnt
    asm volatile("global_load_async_to_lds_b128 %0, %1, off"
                 :: "v"(l0), "v"((unsigned long long)ga) : "memory");
    asm volatile("global_load_async_to_lds_b128 %0, %1, off"
                 :: "v"(l1), "v"((unsigned long long)(ga + (size_t)64 * DIM)) : "memory");
  };
  auto issueB = [&](int k0) {
#pragma unroll
    for (int j = 0; j < 8; j++) {
      int p = bk + 2 * j;
      const float* wp = W + (size_t)(k0 + 2 * p) * HID + (n0 + bn);
      bReg[j] = make_float2(wp[0], wp[HID]);
    }
  };
  auto storeB = [&](int buf) {
#pragma unroll
    for (int j = 0; j < 8; j++) {
      int p = bk + 2 * j;
      *(unsigned*)&sB[buf][bn * AP + 2 * p] = packbf2(bReg[j].x, bReg[j].y);
    }
  };

  const v8f vzero = {0.f, 0.f, 0.f, 0.f, 0.f, 0.f, 0.f, 0.f};
  v8f acc[4][2];
#pragma unroll
  for (int m = 0; m < 4; m++)
#pragma unroll
    for (int n = 0; n < 2; n++) acc[m][n] = vzero;

  const int fr = lane & 15;
  const int kA = (lane >> 4) * 8;
  const int kB = (lane >> 4) * 8;

  auto compute = [&](int buf) {
    v16bf bf[2];
#pragma unroll
    for (int n = 0; n < 2; n++)
      bf[n] = ld_frag(&sB[buf][(wn * 32 + n * 16 + fr) * AP + kB], 16);
#pragma unroll
    for (int m = 0; m < 4; m++) {
      v16bf a = ld_frag(&sA[buf][(wm * 64 + m * 16 + fr) * AP + kA], 16);
#pragma unroll
      for (int n = 0; n < 2; n++)
        acc[m][n] = wmma_bf16(a, bf[n], acc[m][n]);
    }
  };

  issueA(0, 0); issueB(0); storeB(0);
  asm volatile("s_wait_asynccnt 0x0" ::: "memory");
  __syncthreads();

  const int nk = DIM / KC;  // 64
  for (int ks = 0; ks < nk; ks++) {
    int buf = ks & 1;
    if (ks + 1 < nk) { issueA((ks + 1) * KC, buf ^ 1); issueB((ks + 1) * KC); }
    compute(buf);
    if (ks + 1 < nk) storeB(buf ^ 1);
    asm volatile("s_wait_asynccnt 0x0" ::: "memory");
    __syncthreads();
  }

  // Epilogue: bias + routing weight; expert 0 initializes the output.
  const float* bias = db + (size_t)e * HID;
#pragma unroll
  for (int m = 0; m < 4; m++) {
    int rbase = t0 + wm * 64 + m * 16 + ((lane >> 4) << 3);
#pragma unroll
    for (int n = 0; n < 2; n++) {
      int hcol = n0 + wn * 32 + n * 16 + fr;
      float bv = bias[hcol];
#pragma unroll
      for (int v = 0; v < 8; v++) {
        int row = rbase + v;
        float wgt = rw[(size_t)row * NE + e];
        float val = wgt * (acc[m][n][v] + bv);
        float* op = out + (size_t)row * HID + hcol;
        if (e == 0) *op = val; else *op += val;
      }
    }
  }
}

extern "C" void kernel_launch(void* const* d_in, const int* in_sizes, int n_in,
                              void* d_out, int out_size, void* d_ws, size_t ws_size,
                              hipStream_t stream) {
  const float* x   = (const float*)d_in[0];   // hidden_states (B,S,H)
  const float* rw  = (const float*)d_in[1];   // routing_weights (T,E)
  const float* gup = (const float*)d_in[2];   // gate_up_proj (E,H,2D)
  const float* gub = (const float*)d_in[3];   // gate_up_proj_bias (E,2D)
  const float* dpw = (const float*)d_in[4];   // down_proj (E,D,H)
  const float* dpb = (const float*)d_in[5];   // down_proj_bias (E,H)
  float* out = (float*)d_out;
  unsigned short* act = (unsigned short*)d_ws;  // TOK*DIM bf16 = 16 MiB scratch

  dim3 blk(256);
  for (int e = 0; e < NE; e++) {
    moe_gateup_bf16<<<dim3(TOK / BT, DIM / BN), blk, 0, stream>>>(x, gup, gub, act, e);
    moe_down_bf16<<<dim3(TOK / BT, HID / BN), blk, 0, stream>>>(act, dpw, dpb, rw, out, e);
  }
  (void)in_sizes; (void)n_in; (void)out_size; (void)ws_size;
}
